// SimpleAttLayer_38852274160086
// MI455X (gfx1250) — matrix operand order repaired and checked by reading, not compiled, not run on hardware
//
#include <hip/hip_runtime.h>

typedef __attribute__((ext_vector_type(16))) __bf16 v16bf;
typedef __attribute__((ext_vector_type(2)))  __bf16 bf16x2;
typedef __attribute__((ext_vector_type(8)))  float  v8f;
typedef __attribute__((ext_vector_type(4)))  unsigned int u32x4;

#define IN_D  512
#define ATT_A 256
#define TLEN  4

__device__ __forceinline__ float fast_exp(float x) {
  return __builtin_amdgcn_exp2f(x * 1.4426950408889634f);
}
__device__ __forceinline__ float fast_tanh(float x) {
#if __has_builtin(__builtin_amdgcn_tanhf)
  return __builtin_amdgcn_tanhf(x);            // native v_tanh_f32 (CDNA5 TRANS op)
#else
  x = fminf(fmaxf(x, -15.f), 15.f);
  float e = __builtin_amdgcn_exp2f(x * 2.8853900817779268f); // exp(2x)
  return (e - 1.f) * __builtin_amdgcn_rcpf(e + 1.f);
#endif
}

union ABv { v16bf v; u32x4 q[2]; };

__global__ void __launch_bounds__(256)
att_kernel(const float* __restrict__ X, const float* __restrict__ W,
           const float* __restrict__ B, const float* __restrict__ U,
           float* __restrict__ out, int N, int SPW) {
  extern __shared__ unsigned char smem[];
  bf16x2* wfrag = (bf16x2*)smem;                        // 256KB swizzled bf16 W fragments
  float* ldsB = (float*)(smem + 262144);
  float* ldsU = (float*)(smem + 262144 + 1024);

  const int tid = threadIdx.x;
  if (tid < ATT_A) { ldsB[tid] = B[tid]; ldsU[tid] = U[tid]; }

  // ---- stage W (f32 global -> bf16 LDS, pre-swizzled into WMMA B-fragment order) ----
  // index t = ((j*16 + kk)*32 + lane)*8 + i ; unit i holds (k0, k0+1) for column n
#pragma unroll 1
  for (int t = tid; t < 16 * 16 * 32 * 8; t += 256) {
    int i  = t & 7;
    int l  = (t >> 3) & 31;
    int kk = (t >> 8) & 15;
    int j  = (t >> 12) & 15;
    int hi = l >> 4;
    int n  = j * 16 + (l & 15);
    int k0 = kk * 32 + hi * 16 + 2 * i;                 // B layout: lanes0-15 K=0-15, lanes16-31 K=16-31
    bf16x2 pr;
    pr.x = (__bf16)W[(size_t)k0 * ATT_A + n];
    pr.y = (__bf16)W[(size_t)(k0 + 1) * ATT_A + n];
    wfrag[t] = pr;
  }
  __syncthreads();

  const int lane = tid & 31;
  const int wave = tid >> 5;
  const int m    = lane & 15;
  const int hi   = lane >> 4;

  const int wgFirst = blockIdx.x * SPW;
  int wgEnd = wgFirst + SPW; if (wgEnd > N) wgEnd = N;
  float* alphaOut = out + (size_t)N * IN_D;

#pragma unroll 1
  for (int s0 = wgFirst + wave * 4; s0 + 4 <= wgEnd; s0 += 32) {
    const int rowBase = s0 * TLEN;                      // 16 contiguous rows = 4 samples
    const float* rowPtr = X + (size_t)(rowBase + m) * IN_D;

    // ---- build A fragments (16 rows x 512 K, bf16), contiguous k = 16h+s order (same as B) ----
    // hardware f32->bf16 converts (v_cvt_pk_bf16_f32)
    v16bf a[16];
#pragma unroll
    for (int kk = 0; kk < 16; ++kk) {
      int kA = kk * 32 + 16 * hi;                       // VGPRs 0..3 : K = kA..kA+7
      int kB = kk * 32 + 16 * hi + 8;                   // VGPRs 4..7 : K = kB..kB+7 (contiguous 16h+s, matches B)
      float4 fa0 = *(const float4*)(rowPtr + kA);
      float4 fa1 = *(const float4*)(rowPtr + kA + 4);
      float4 fb0 = *(const float4*)(rowPtr + kB);
      float4 fb1 = *(const float4*)(rowPtr + kB + 4);
      v16bf t;
      t[0]  = (__bf16)fa0.x; t[1]  = (__bf16)fa0.y;
      t[2]  = (__bf16)fa0.z; t[3]  = (__bf16)fa0.w;
      t[4]  = (__bf16)fa1.x; t[5]  = (__bf16)fa1.y;
      t[6]  = (__bf16)fa1.z; t[7]  = (__bf16)fa1.w;
      t[8]  = (__bf16)fb0.x; t[9]  = (__bf16)fb0.y;
      t[10] = (__bf16)fb0.z; t[11] = (__bf16)fb0.w;
      t[12] = (__bf16)fb1.x; t[13] = (__bf16)fb1.y;
      t[14] = (__bf16)fb1.z; t[15] = (__bf16)fb1.w;
      a[kk] = t;
    }

    float p[8];
#pragma unroll
    for (int v = 0; v < 8; ++v) p[v] = 0.f;

    // ---- GEMM sweep: 16 column tiles x 16 K-steps, dual accumulators ----
    // keep this loop ROLLED: full unroll spills B fragments to scratch
#pragma unroll 1
    for (int j = 0; j < 16; ++j) {
      // one base pointer; all kk strides become ds_load_b128 immediate offsets
      const u32x4* basev = (const u32x4*)((unsigned char*)smem + j * 16384 + lane * 32);
      v8f c0 = {}; v8f c1 = {};
#pragma unroll
      for (int kk = 0; kk < 16; kk += 2) {
        ABv b0, b1;
        b0.q[0] = basev[kk * 64 + 0];                   // byte offset kk*1024
        b0.q[1] = basev[kk * 64 + 1];                   // byte offset kk*1024+16
        b1.q[0] = basev[kk * 64 + 64];                  // byte offset (kk+1)*1024
        b1.q[1] = basev[kk * 64 + 65];
        c0 = __builtin_amdgcn_wmma_f32_16x16x32_bf16(false, a[kk],     false, b0.v, (short)0, c0, false, false);
        c1 = __builtin_amdgcn_wmma_f32_16x16x32_bf16(false, a[kk + 1], false, b1.v, (short)0, c1, false, false);
      }
      int n = j * 16 + m;
      float bn = ldsB[n], un = ldsU[n];
#pragma unroll
      for (int v = 0; v < 8; ++v)
        p[v] += fast_tanh(c0[v] + c1[v] + bn) * un;     // row M = v + 8*hi, col n
    }

    // ---- reduce per-row scores across the 16 column-lanes ----
#pragma unroll
    for (int v = 0; v < 8; ++v) {
      p[v] += __shfl_xor(p[v], 1, 32);
      p[v] += __shfl_xor(p[v], 2, 32);
      p[v] += __shfl_xor(p[v], 4, 32);
      p[v] += __shfl_xor(p[v], 8, 32);
    }

    // ---- softmax over T=4 (rows hi*8+v => samples 2hi, 2hi+1 in this half) ----
    float a8[8];
#pragma unroll
    for (int g = 0; g < 2; ++g) {
      float mx = fmaxf(fmaxf(p[4*g], p[4*g+1]), fmaxf(p[4*g+2], p[4*g+3]));
      float e0 = fast_exp(p[4*g]   - mx);
      float e1 = fast_exp(p[4*g+1] - mx);
      float e2 = fast_exp(p[4*g+2] - mx);
      float e3 = fast_exp(p[4*g+3] - mx);
      float r = __builtin_amdgcn_rcpf(e0 + e1 + e2 + e3);
      a8[4*g] = e0*r; a8[4*g+1] = e1*r; a8[4*g+2] = e2*r; a8[4*g+3] = e3*r;
    }
    float a8o[8];
#pragma unroll
    for (int i = 0; i < 8; ++i) a8o[i] = __shfl_xor(a8[i], 16, 32);
    float alp[16];
#pragma unroll
    for (int r = 0; r < 8; ++r) {
      alp[r]     = hi ? a8o[r] : a8[r];
      alp[r + 8] = hi ? a8[r]  : a8o[r];
    }

    // ---- alphas out (lane 0 writes all 16 as 4x float4) ----
    if (lane == 0) {
      float4 q;
      q.x = alp[0];  q.y = alp[1];  q.z = alp[2];  q.w = alp[3];  *(float4*)(alphaOut + rowBase)      = q;
      q.x = alp[4];  q.y = alp[5];  q.z = alp[6];  q.w = alp[7];  *(float4*)(alphaOut + rowBase + 4)  = q;
      q.x = alp[8];  q.y = alp[9];  q.z = alp[10]; q.w = alp[11]; *(float4*)(alphaOut + rowBase + 8)  = q;
      q.x = alp[12]; q.y = alp[13]; q.z = alp[14]; q.w = alp[15]; *(float4*)(alphaOut + rowBase + 12) = q;
    }

    // ---- fp32 weighted sum: out[s] = sum_t alpha[t] * X[s,t,:]  (re-reads hit L2) ----
#pragma unroll
    for (int s = 0; s < 4; ++s) {
      const float* r0 = X + (size_t)(rowBase + 4 * s) * IN_D;
      float* o0 = out + (size_t)(s0 + s) * IN_D;
      float a0 = alp[4*s], a1 = alp[4*s+1], a2 = alp[4*s+2], a3 = alp[4*s+3];
#pragma unroll
      for (int ch = 0; ch < 4; ++ch) {
        int col = ch * 128 + lane * 4;
        float4 x0 = *(const float4*)(r0 + col);
        float4 x1 = *(const float4*)(r0 + IN_D + col);
        float4 x2 = *(const float4*)(r0 + 2 * IN_D + col);
        float4 x3 = *(const float4*)(r0 + 3 * IN_D + col);
        float4 o;
        o.x = a0*x0.x + a1*x1.x + a2*x2.x + a3*x3.x;
        o.y = a0*x0.y + a1*x1.y + a2*x2.y + a3*x3.y;
        o.z = a0*x0.z + a1*x1.z + a2*x2.z + a3*x3.z;
        o.w = a0*x0.w + a1*x1.w + a2*x2.w + a3*x3.w;
        *(float4*)(o0 + col) = o;
      }
    }
  }
}

extern "C" void kernel_launch(void* const* d_in, const int* in_sizes, int n_in,
                              void* d_out, int out_size, void* d_ws, size_t ws_size,
                              hipStream_t stream) {
  const float* X = (const float*)d_in[0];
  const float* W = (const float*)d_in[1];
  const float* B = (const float*)d_in[2];
  const float* U = (const float*)d_in[3];
  float* out = (float*)d_out;

  int N = in_sizes[0] / (TLEN * IN_D);      // 131072
  int SPW = 256;                            // samples per workgroup (8 iterations of 8 waves x 4)
  int nWG = (N + SPW - 1) / SPW;            // 512
  size_t smemBytes = 262144 + 2048;         // swizzled W (bf16) + b + u

  att_kernel<<<nWG, 256, smemBytes, stream>>>(X, W, B, U, out, N, SPW);
}
